// LSTMv1_44573170598543
// MI455X (gfx1250) — matrix operand order repaired and checked by reading, not compiled, not run on hardware
//
#include <hip/hip_runtime.h>
#include <math.h>

// ---------------------------------------------------------------------------
// Residual LSTM stack for MI455X (gfx1250, wave32, WMMA).
// bf16 WMMA GEMMs with fp32 accumulate; weights packed once per call into the
// wave32 B-matrix lane layout; activations kept in the A-matrix lane layout.
// Each GEMM wave is M-blocked 4x (4 D tiles) so every weight fragment fetched
// from L2 feeds 4 WMMAs -> ~4x less L2 weight traffic.
// ---------------------------------------------------------------------------

typedef __attribute__((ext_vector_type(16))) __bf16 v16bf;
typedef __attribute__((ext_vector_type(8)))  float  v8f;

#define HDIM   512
#define H4DIM  2048
#define BATCH  256
#define NBLK   4
#define TSTEPS 256
#define LN_EPS 1e-5f

// ---- lane-layout helpers (CDNA5 ISA 7.12.2, 16-bit matrices, wave32) -------
__device__ __forceinline__ int apack_pos(int m, int h) {
    int kt   = h >> 5;
    int k32  = h & 31;
    int lane = (m & 15) + (((k32 >> 3) & 1) << 4);
    int e    = (k32 & 7) + ((k32 >> 4) << 3);
    return (kt * 32 + lane) * 16 + e;
}

__device__ __forceinline__ float sigmoid_f(float x) {
    return 1.0f / (1.0f + __expf(-x));
}
__device__ __forceinline__ float gelu_exact(float x) {
    return 0.5f * x * (1.0f + erff(x * 0.70710678118654752f));
}

// ---------------------------------------------------------------------------
// Weight packing into B-matrix tiles:
// dst[((ntile*KT + kt)*32 + lane)*16 + e] = W[n][k],
// n = ntile*16 + (lane&15), k = kt*32 + ((e>=8)?16:0) + ((lane>=16)?8:0) + (e&7).
// ---------------------------------------------------------------------------
__global__ void pack_w_kernel(const float* __restrict__ src, __bf16* __restrict__ dst,
                              int ktiles, int total) {
    int d = blockIdx.x * blockDim.x + threadIdx.x;
    if (d >= total) return;
    int per_nt = ktiles << 9;
    int ntile  = d / per_nt;
    int r      = d - ntile * per_nt;
    int kt     = r >> 9;
    int r2     = r & 511;
    int lane   = r2 >> 4;
    int e      = r2 & 15;
    int n      = ntile * 16 + (lane & 15);
    int K      = ktiles << 5;
    int k      = kt * 32 + ((e >> 3) << 4) + ((lane >> 4) << 3) + (e & 7);
    dst[d] = (__bf16)src[n * K + k];
}

__global__ void zero_state_kernel(float* __restrict__ c, __bf16* __restrict__ hpk, int n) {
    int i = blockIdx.x * blockDim.x + threadIdx.x;
    if (i < n) { c[i] = 0.0f; hpk[i] = (__bf16)0.0f; }
}

// ---------------------------------------------------------------------------
// Embedding concat -> x [B,H] fp32 for timestep t.
// ---------------------------------------------------------------------------
__global__ void embed_kernel(const float* __restrict__ gs,
                             const int* __restrict__ stage, const int* __restrict__ ec,
                             const int* __restrict__ oc, const int* __restrict__ ea,
                             const int* __restrict__ oa,
                             const float* __restrict__ emb_s, const float* __restrict__ emb_c,
                             const float* __restrict__ emb_a,
                             float* __restrict__ x, int t) {
    int idx = blockIdx.x * blockDim.x + threadIdx.x;
    int b = idx >> 9, h = idx & 511;
    int bt = b * TSTEPS + t;
    float v;
    if      (h < 64)  v = emb_s[stage[bt] * 64 + h];
    else if (h < 128) v = emb_c[ec[bt] * 64 + (h - 64)];
    else if (h < 192) v = emb_c[oc[bt] * 64 + (h - 128)];
    else if (h < 320) v = emb_a[ea[bt] * 128 + (h - 192)];
    else if (h < 448) v = emb_a[oa[bt] * 128 + (h - 320)];
    else              v = gs[bt * 64 + (h - 448)];
    x[idx] = v;
}

// ---------------------------------------------------------------------------
// LayerNorm over H=512 + scatter into packed A layout (bf16).
// Grid: 16 blocks (one per 16-row group) x 256 threads (16 threads per row).
// ---------------------------------------------------------------------------
__global__ void ln_pack_kernel(const float* __restrict__ x,
                               const float* __restrict__ gain, const float* __restrict__ bias,
                               __bf16* __restrict__ dst) {
    __shared__ float s1[256], s2[256];
    __shared__ float smean[16], srstd[16];
    int t = threadIdx.x, m = t >> 4, p = t & 15;
    int row = blockIdx.x * 16 + m;
    const float* xr = x + row * HDIM;
    float a = 0.f, q = 0.f;
    #pragma unroll
    for (int j = 0; j < 32; ++j) {
        float v = xr[p + (j << 4)];
        a += v; q += v * v;
    }
    s1[t] = a; s2[t] = q;
    __syncthreads();
    if (p == 0) {
        float A = 0.f, Q = 0.f;
        #pragma unroll
        for (int j = 0; j < 16; ++j) { A += s1[(m << 4) + j]; Q += s2[(m << 4) + j]; }
        float mean = A * (1.0f / HDIM);
        float var  = Q * (1.0f / HDIM) - mean * mean;
        smean[m] = mean;
        srstd[m] = rsqrtf(var + LN_EPS);
    }
    __syncthreads();
    float mean = smean[m], rstd = srstd[m];
    __bf16* drg = dst + blockIdx.x * (16 * HDIM);
    #pragma unroll
    for (int j = 0; j < 32; ++j) {
        int h = p + (j << 4);
        float v = (xr[h] - mean) * rstd * gain[h] + bias[h];
        drg[apack_pos(m, h)] = (__bf16)v;
    }
}

// ---------------------------------------------------------------------------
// gates[B,2048] = xn @ Wih^T + h @ Whh^T.
// Grid: dim3(16 colgroups, 4 rowquads), 256 threads = 8 waves.
// Each wave: 1 N tile x 4 M tiles; each B fragment feeds 4 WMMAs.
// ---------------------------------------------------------------------------
__global__ void gates_gemm_kernel(const __bf16* __restrict__ xnpk,
                                  const __bf16* __restrict__ hpk,
                                  const __bf16* __restrict__ wih,
                                  const __bf16* __restrict__ whh,
                                  float* __restrict__ gates) {
    int lane  = threadIdx.x & 31;
    int wv    = threadIdx.x >> 5;
    int rq    = blockIdx.y;                    // row quad: 4 M tiles
    int ntile = blockIdx.x * 8 + wv;

    const v16bf* A  = (const v16bf*)xnpk + (rq * 4 * 16) * 32 + lane;
    const v16bf* Ah = (const v16bf*)hpk  + (rq * 4 * 16) * 32 + lane;
    const v16bf* B1 = (const v16bf*)wih  + (ntile * 16) * 32 + lane;
    const v16bf* B2 = (const v16bf*)whh  + (ntile * 16) * 32 + lane;

    v8f acc[4] = {};
    #pragma unroll 2
    for (int kt = 0; kt < 16; ++kt) {
        v16bf b1 = B1[kt * 32];
        v16bf b2 = B2[kt * 32];
        #pragma unroll
        for (int r = 0; r < 4; ++r) {
            v16bf a  = A [(r * 16 + kt) * 32];
            acc[r] = __builtin_amdgcn_wmma_f32_16x16x32_bf16(false, a, false, b1,
                                                             (short)0, acc[r], false, false);
            v16bf ah = Ah[(r * 16 + kt) * 32];
            acc[r] = __builtin_amdgcn_wmma_f32_16x16x32_bf16(false, ah, false, b2,
                                                             (short)0, acc[r], false, false);
        }
    }
    int n0  = ntile * 16 + (lane & 15);
    int mhi = (lane >> 4) << 3;
    #pragma unroll
    for (int r = 0; r < 4; ++r) {
        int m0 = (rq * 4 + r) * 16 + mhi;
        #pragma unroll
        for (int v = 0; v < 8; ++v)
            gates[(m0 + v) * H4DIM + n0] = acc[r][v];
    }
}

// ---------------------------------------------------------------------------
// LSTM pointwise + residual + h repack.
// ---------------------------------------------------------------------------
__global__ void lstm_ew_kernel(const float* __restrict__ gates,
                               const float* __restrict__ bih, const float* __restrict__ bhh,
                               const float* __restrict__ x,
                               float* __restrict__ c, __bf16* __restrict__ hpk,
                               float* __restrict__ y) {
    int idx = blockIdx.x * blockDim.x + threadIdx.x;
    int b = idx >> 9, h = idx & 511;
    const float* gr = gates + b * H4DIM;
    float ig = gr[h]        + bih[h]        + bhh[h];
    float fg = gr[h + 512]  + bih[h + 512]  + bhh[h + 512];
    float gg = gr[h + 1024] + bih[h + 1024] + bhh[h + 1024];
    float og = gr[h + 1536] + bih[h + 1536] + bhh[h + 1536];
    float cn = sigmoid_f(fg) * c[idx] + sigmoid_f(ig) * tanhf(gg);
    float hn = sigmoid_f(og) * tanhf(cn);
    c[idx] = cn;
    int rg = b >> 4, m = b & 15;
    hpk[rg * (16 * HDIM) + apack_pos(m, h)] = (__bf16)hn;
    y[idx] = x[idx] + hn;
}

// ---------------------------------------------------------------------------
// mid = gelu(yn @ fc_w^T + fc_b), written in packed A layout (K=2048).
// Grid: dim3(16, 4), 8 waves; 4 M tiles per wave.
// ---------------------------------------------------------------------------
__global__ void fc_gemm_kernel(const __bf16* __restrict__ ynpk,
                               const __bf16* __restrict__ fcw,
                               const float* __restrict__ fcb,
                               __bf16* __restrict__ midpk) {
    int lane  = threadIdx.x & 31;
    int wv    = threadIdx.x >> 5;
    int rq    = blockIdx.y;
    int ntile = blockIdx.x * 8 + wv;

    const v16bf* A  = (const v16bf*)ynpk + (rq * 4 * 16) * 32 + lane;
    const v16bf* B1 = (const v16bf*)fcw  + (ntile * 16) * 32 + lane;

    v8f acc[4] = {};
    #pragma unroll 2
    for (int kt = 0; kt < 16; ++kt) {
        v16bf b = B1[kt * 32];
        #pragma unroll
        for (int r = 0; r < 4; ++r) {
            v16bf a = A[(r * 16 + kt) * 32];
            acc[r] = __builtin_amdgcn_wmma_f32_16x16x32_bf16(false, a, false, b,
                                                             (short)0, acc[r], false, false);
        }
    }
    int nloc = lane & 15;
    int ncol = ntile * 16 + nloc;
    int mhi  = (lane >> 4) << 3;
    float bias = fcb[ncol];
    #pragma unroll
    for (int r = 0; r < 4; ++r) {
        __bf16* drg = midpk + (rq * 4 + r) * (16 * H4DIM);
        #pragma unroll
        for (int v = 0; v < 8; ++v) {
            float g = gelu_exact(acc[r][v] + bias);
            drg[apack_pos(mhi + v, ncol)] = (__bf16)g;
        }
    }
}

// ---------------------------------------------------------------------------
// x' = y + mid @ proj_w^T + proj_b.  N=512 (32 ntiles), K=2048 (64 ktiles).
// Grid: dim3(4, 4), 8 waves; 4 M tiles per wave.
// ---------------------------------------------------------------------------
__global__ void proj_gemm_kernel(const __bf16* __restrict__ midpk,
                                 const __bf16* __restrict__ projw,
                                 const float* __restrict__ projb,
                                 const float* __restrict__ y,
                                 float* __restrict__ x) {
    int lane  = threadIdx.x & 31;
    int wv    = threadIdx.x >> 5;
    int rq    = blockIdx.y;
    int ntile = blockIdx.x * 8 + wv;

    const v16bf* A  = (const v16bf*)midpk + (rq * 4 * 64) * 32 + lane;
    const v16bf* B1 = (const v16bf*)projw + (ntile * 64) * 32 + lane;

    v8f acc[4] = {};
    #pragma unroll 2
    for (int kt = 0; kt < 64; ++kt) {
        v16bf b = B1[kt * 32];
        #pragma unroll
        for (int r = 0; r < 4; ++r) {
            v16bf a = A[(r * 64 + kt) * 32];
            acc[r] = __builtin_amdgcn_wmma_f32_16x16x32_bf16(false, a, false, b,
                                                             (short)0, acc[r], false, false);
        }
    }
    int n0  = ntile * 16 + (lane & 15);
    int mhi = (lane >> 4) << 3;
    float bias = projb[n0];
    #pragma unroll
    for (int r = 0; r < 4; ++r) {
        int m0 = (rq * 4 + r) * 16 + mhi;
        #pragma unroll
        for (int v = 0; v < 8; ++v) {
            int row = m0 + v;
            x[row * HDIM + n0] = acc[r][v] + bias + y[row * HDIM + n0];
        }
    }
}

// ---------------------------------------------------------------------------
// Output heads on the final timestep (58x512 per row, tiny).
// ---------------------------------------------------------------------------
__global__ void heads_kernel(const float* __restrict__ x,
                             const float* __restrict__ bw, const float* __restrict__ bb,
                             const float* __restrict__ mw, const float* __restrict__ mb,
                             const float* __restrict__ cw, const float* __restrict__ cb,
                             float* __restrict__ out) {
    __shared__ float red[256];
    int b = blockIdx.x, t = threadIdx.x;
    const float* xr = x + b * HDIM;
    for (int o = 0; o < 58; ++o) {
        const float* wrow; float bias;
        if      (o < 16) { wrow = bw + o * HDIM;        bias = bb[o]; }
        else if (o < 37) { wrow = mw + (o - 16) * HDIM; bias = mb[o - 16]; }
        else             { wrow = cw + (o - 37) * HDIM; bias = cb[o - 37]; }
        float s = 0.f;
        for (int j = t; j < HDIM; j += 256) s += xr[j] * wrow[j];
        red[t] = s;
        __syncthreads();
        for (int st = 128; st > 0; st >>= 1) {
            if (t < st) red[t] += red[t + st];
            __syncthreads();
        }
        if (t == 0) {
            int pos = (o < 16) ? b * 16 + o
                    : (o < 37) ? 4096 + b * 21 + (o - 16)
                               : 4096 + 5376 + b * 21 + (o - 37);
            out[pos] = red[0] + bias;
        }
        __syncthreads();
    }
}

// ---------------------------------------------------------------------------
extern "C" void kernel_launch(void* const* d_in, const int* in_sizes, int n_in,
                              void* d_out, int out_size, void* d_ws, size_t ws_size,
                              hipStream_t stream) {
    const float* gamestate = (const float*)d_in[0];
    const int*   stage     = (const int*)d_in[1];
    const int*   ego_char  = (const int*)d_in[2];
    const int*   opp_char  = (const int*)d_in[3];
    const int*   ego_act   = (const int*)d_in[4];
    const int*   opp_act   = (const int*)d_in[5];
    const float* emb_stage = (const float*)d_in[6];
    const float* emb_char  = (const float*)d_in[7];
    const float* emb_act   = (const float*)d_in[8];
    const float* ln1_g     = (const float*)d_in[9];
    const float* ln1_b     = (const float*)d_in[10];
    const float* w_ih      = (const float*)d_in[11];
    const float* w_hh      = (const float*)d_in[12];
    const float* b_ih      = (const float*)d_in[13];
    const float* b_hh      = (const float*)d_in[14];
    const float* ln2_g     = (const float*)d_in[15];
    const float* ln2_b     = (const float*)d_in[16];
    const float* fc_w      = (const float*)d_in[17];
    const float* fc_b      = (const float*)d_in[18];
    const float* proj_w    = (const float*)d_in[19];
    const float* proj_b    = (const float*)d_in[20];
    const float* btn_w     = (const float*)d_in[21];
    const float* btn_b     = (const float*)d_in[22];
    const float* ms_w      = (const float*)d_in[23];
    const float* ms_b      = (const float*)d_in[24];
    const float* cs_w      = (const float*)d_in[25];
    const float* cs_b      = (const float*)d_in[26];

    char* ws = (char*)d_ws;
    size_t off = 0;
    auto take = [&](size_t bytes) -> void* {
        off = (off + 255) & ~(size_t)255;
        void* p = ws + off;
        off += bytes;
        return p;
    };
    const size_t WMAT = (size_t)NBLK * H4DIM * HDIM;
    __bf16* wihp  = (__bf16*)take(WMAT * 2);
    __bf16* whhp  = (__bf16*)take(WMAT * 2);
    __bf16* fcp   = (__bf16*)take(WMAT * 2);
    __bf16* projp = (__bf16*)take((size_t)NBLK * HDIM * H4DIM * 2);
    __bf16* hpk   = (__bf16*)take((size_t)NBLK * BATCH * HDIM * 2);
    float*  cst   = (float*) take((size_t)NBLK * BATCH * HDIM * 4);
    float*  xbuf  = (float*) take((size_t)BATCH * HDIM * 4);
    float*  ybuf  = (float*) take((size_t)BATCH * HDIM * 4);
    __bf16* xnpk  = (__bf16*)take((size_t)BATCH * HDIM * 2);
    __bf16* ynpk  = (__bf16*)take((size_t)BATCH * HDIM * 2);
    float*  gates = (float*) take((size_t)BATCH * H4DIM * 4);
    __bf16* midpk = (__bf16*)take((size_t)BATCH * H4DIM * 2);
    (void)ws_size; (void)in_sizes; (void)n_in; (void)out_size;

    const int MATE = H4DIM * HDIM;

    for (int i = 0; i < NBLK; ++i) {
        pack_w_kernel<<<4096, 256, 0, stream>>>(w_ih + (size_t)i * MATE, wihp + (size_t)i * MATE, 16, MATE);
        pack_w_kernel<<<4096, 256, 0, stream>>>(w_hh + (size_t)i * MATE, whhp + (size_t)i * MATE, 16, MATE);
        pack_w_kernel<<<4096, 256, 0, stream>>>(fc_w + (size_t)i * MATE, fcp  + (size_t)i * MATE, 16, MATE);
        pack_w_kernel<<<4096, 256, 0, stream>>>(proj_w + (size_t)i * MATE, projp + (size_t)i * MATE, 64, MATE);
    }
    zero_state_kernel<<<2048, 256, 0, stream>>>(cst, hpk, NBLK * BATCH * HDIM);

    const int BH = BATCH * HDIM;
    dim3 g2048(16, 4), g512(4, 4);

    for (int t = 0; t < TSTEPS; ++t) {
        embed_kernel<<<512, 256, 0, stream>>>(gamestate, stage, ego_char, opp_char,
                                              ego_act, opp_act, emb_stage, emb_char,
                                              emb_act, xbuf, t);
        for (int i = 0; i < NBLK; ++i) {
            ln_pack_kernel<<<16, 256, 0, stream>>>(xbuf, ln1_g + i * HDIM, ln1_b + i * HDIM, xnpk);
            gates_gemm_kernel<<<g2048, 256, 0, stream>>>(xnpk, hpk + (size_t)i * BH,
                                                         wihp + (size_t)i * MATE,
                                                         whhp + (size_t)i * MATE, gates);
            lstm_ew_kernel<<<512, 256, 0, stream>>>(gates, b_ih + i * H4DIM, b_hh + i * H4DIM,
                                                    xbuf, cst + (size_t)i * BH,
                                                    hpk + (size_t)i * BH, ybuf);
            ln_pack_kernel<<<16, 256, 0, stream>>>(ybuf, ln2_g + i * HDIM, ln2_b + i * HDIM, ynpk);
            fc_gemm_kernel<<<g2048, 256, 0, stream>>>(ynpk, fcp + (size_t)i * MATE,
                                                      fc_b + i * H4DIM, midpk);
            proj_gemm_kernel<<<g512, 256, 0, stream>>>(midpk, projp + (size_t)i * MATE,
                                                       proj_b + i * HDIM, ybuf, xbuf);
        }
    }
    heads_kernel<<<256, 256, 0, stream>>>(xbuf, btn_w, btn_b, ms_w, ms_b, cs_w, cs_b,
                                          (float*)d_out);
}
